// DecoderBlock_87436944212043
// MI455X (gfx1250) — matrix operand order repaired and checked by reading, not compiled, not run on hardware
//
#include <hip/hip_runtime.h>
#include <hip/hip_bf16.h>

// Problem constants (match reference).
#define LNUM 4
#define VNUM 32000
#define DNUM 512
#define HNUM 8
#define SNUM 1024
#define FNUM 2048
#define DHNUM 64
#define BNUM 4
#define MNUM (BNUM * SNUM)   // 4096 token rows

typedef __bf16 bf16_t;
typedef __attribute__((ext_vector_type(16))) __bf16 v16bf;
typedef __attribute__((ext_vector_type(8)))  __bf16 v8bf;
typedef __attribute__((ext_vector_type(8)))  float  v8f;
typedef __attribute__((ext_vector_type(4)))  unsigned int u32x4;
typedef __attribute__((ext_vector_type(8)))  int i32x8;
typedef __attribute__((ext_vector_type(4)))  int i32x4;

#if defined(__gfx1250__) && __has_builtin(__builtin_amdgcn_tensor_load_to_lds)
#define USE_TDM 1
#else
#define USE_TDM 0
#endif

// ---------------------------------------------------------------------------
// WMMA helpers (CDNA5 wave32, V_WMMA_F32_16X16X32_BF16)
// ---------------------------------------------------------------------------
static __device__ __forceinline__ v8f wmma_bf16(v16bf a, v16bf b, v8f c) {
  // (neg_a, A, neg_b, B, c_mod, C, reuse_a, reuse_b)
  return __builtin_amdgcn_wmma_f32_16x16x32_bf16(false, a, false, b,
                                                 (short)0, c, false, false);
}

// A-operand: 16x32 (MxK). lane holds row = lane&15; elements 0..15 = contiguous K chunk at (lane>=16)*16.
static __device__ __forceinline__ v16bf load_a_tile(const bf16_t* base, int stride, int lane) {
  int r  = lane & 15;
  int k0 = (lane >> 4) * 16;
  return *(const v16bf*)(base + (size_t)r * stride + k0);
}

// B-operand: 32x16 (KxN) sourced from transposed weight Wt[N][K] (row-major).
// lane holds column n = lane&15; elements 0..15 = contiguous K chunk at (lane>=16)*16.
static __device__ __forceinline__ v16bf load_b_tile(const bf16_t* base, int stride, int lane) {
  int n  = lane & 15;
  int k0 = (lane >> 4) * 16;
  return *(const v16bf*)(base + (size_t)n * stride + k0);
}

// ---------------------------------------------------------------------------
// Embedding + sinusoidal positional encoding -> f32 + bf16 activations
// ---------------------------------------------------------------------------
__global__ void __launch_bounds__(256)
embed_kernel(const int* __restrict__ tokens, const float* __restrict__ emb,
             float* __restrict__ xf, bf16_t* __restrict__ xb) {
  size_t i = (size_t)blockIdx.x * blockDim.x + threadIdx.x;  // over MNUM*DNUM
  int row = (int)(i / DNUM);
  int d   = (int)(i % DNUM);
  int pos = row % SNUM;
  int tok = tokens[row];
  float div = __expf((float)(d & ~1) * (-logf(10000.0f) / (float)DNUM));
  float ang = (float)pos * div;
  float pe  = (d & 1) ? __cosf(ang) : __sinf(ang);
  float v   = emb[(size_t)tok * DNUM + d] + pe;
  xf[i] = v;
  xb[i] = (bf16_t)v;
}

// ---------------------------------------------------------------------------
// Weight convert + transpose: W[K][N] f32  ->  Wt[N][K] bf16
// ---------------------------------------------------------------------------
__global__ void __launch_bounds__(256)
convert_wt_kernel(const float* __restrict__ W, bf16_t* __restrict__ Wt, int K, int N) {
  size_t i = (size_t)blockIdx.x * blockDim.x + threadIdx.x;
  if (i >= (size_t)K * N) return;
  int n = (int)(i / K);
  int k = (int)(i % K);
  Wt[i] = (bf16_t)W[(size_t)k * N + n];
}

// ---------------------------------------------------------------------------
// WMMA GEMM: C[M,N] = act(A[M,K](bf16) @ Wt[N,K]^T + bias[N])
// One wave computes a 32x64 tile (2 M-tiles x 4 N-tiles, B operands shared);
// block = (32,4) waves; grid = (M/32, N/256).
// ---------------------------------------------------------------------------
template <bool OUT_BF16, bool RELU>
__global__ void __launch_bounds__(128)
gemm_bf16_wmma(const bf16_t* __restrict__ A, const bf16_t* __restrict__ Bt,
               const float* __restrict__ bias, float* __restrict__ Cf,
               bf16_t* __restrict__ Cb, int N, int K) {
  int lane = threadIdx.x;
  int mt   = blockIdx.x;                      // 32-row tile index
  int n64  = blockIdx.y * 4 + threadIdx.y;
  int n0   = n64 * 64;

  const bf16_t* A0 = A + (size_t)mt * 32 * K;
  const bf16_t* A1 = A0 + (size_t)16 * K;
  v8f acc0[4] = {};
  v8f acc1[4] = {};

  for (int k0 = 0; k0 < K; k0 += 32) {
    if (k0 + 32 < K) {  // prefetch next K-step of A (global_prefetch_b8)
      __builtin_prefetch((const void*)(A0 + ((size_t)(lane & 15)) * K + k0 + 32), 0, 0);
    }
    v16bf a0 = load_a_tile(A0 + k0, K, lane);
    v16bf a1 = load_a_tile(A1 + k0, K, lane);
#pragma unroll
    for (int nt = 0; nt < 4; ++nt) {
      v16bf b = load_b_tile(Bt + (size_t)(n0 + nt * 16) * K + k0, K, lane);
      acc0[nt] = wmma_bf16(a0, b, acc0[nt]);
      acc1[nt] = wmma_bf16(a1, b, acc1[nt]);
    }
  }

  // C layout: VGPR j holds row (j + 8*(lane>=16)), column lane&15.
  int col = lane & 15;
  int rh  = (lane >> 4) * 8;
#pragma unroll
  for (int half = 0; half < 2; ++half) {
    v8f* acc = half ? acc1 : acc0;
#pragma unroll
    for (int nt = 0; nt < 4; ++nt) {
#pragma unroll
      for (int j = 0; j < 8; ++j) {
        size_t row = (size_t)mt * 32 + half * 16 + j + rh;
        int c = n0 + nt * 16 + col;
        float v = acc[nt][j] + bias[c];
        if (RELU) v = fmaxf(v, 0.0f);
        if (OUT_BF16) Cb[row * N + c] = (bf16_t)v;
        else          Cf[row * N + c] = v;
      }
    }
  }
}

// ---------------------------------------------------------------------------
// Causal multi-head attention, flash-style online softmax, one wave per
// (batch, head, 16-query tile).  q/k/v are bf16 [M, D]; out is f32 [M, D].
// V block staging uses the Tensor Data Mover (TENSOR_LOAD_TO_LDS) when the
// builtin is available; ordering via s_wait_dscnt / s_wait_tensorcnt.
// ---------------------------------------------------------------------------
__global__ void __launch_bounds__(32)
attn_kernel(const bf16_t* __restrict__ q, const bf16_t* __restrict__ k,
            const bf16_t* __restrict__ v, float* __restrict__ out) {
  __shared__ __align__(16) float  sblk[16][32];   // raw scores for one key block
  __shared__ __align__(16) bf16_t pblk[16][32];   // softmax'd P (A-operand source)
  __shared__ __align__(16) bf16_t vblk[32][64];   // staged V block
  __shared__ float mrow[16], lrow[16], arow[16];

  const float scale = 0.125f;  // 1/sqrt(64)
  int lane = threadIdx.x;
  int qt   = blockIdx.x;        // 0..S/16-1
  int bh   = blockIdx.y;        // 0..B*H-1
  int b    = bh / HNUM;
  int h    = bh % HNUM;
  int mbase = b * SNUM;

  const bf16_t* qh = q + (size_t)mbase * DNUM + h * DHNUM;
  const bf16_t* kh = k + (size_t)mbase * DNUM + h * DHNUM;
  const bf16_t* vh = v + (size_t)mbase * DNUM + h * DHNUM;

  // Q tile A-operands, K(feature)=64 split into 2x32.
  v16bf qa0 = load_a_tile(qh + (size_t)qt * 16 * DNUM + 0,  DNUM, lane);
  v16bf qa1 = load_a_tile(qh + (size_t)qt * 16 * DNUM + 32, DNUM, lane);

  v8f o[4] = {};
  if (lane < 16) { mrow[lane] = -1e30f; lrow[lane] = 0.0f; }

  int qmax = qt * 16 + 15;
  int nkb  = qmax / 32 + 1;  // 32-key blocks (last may be partially masked)

  for (int kb = 0; kb < nkb; ++kb) {
    int kbase = kb * 32;

#if USE_TDM
    // ---- TDM: DMA V tile (32 keys x 64 dims, 2B elems) global -> LDS ----
    {
      // WAR: previous iteration's LDS reads of vblk must be complete before
      // the TDM overwrites it (TDM is unordered vs DS ops).
      asm volatile("s_wait_dscnt 0" ::: "memory");
      unsigned lds_addr = (unsigned)(unsigned long long)(uintptr_t)&vblk[0][0];
      unsigned long long gaddr =
          (unsigned long long)(uintptr_t)(vh + (size_t)kbase * DNUM);
      unsigned tdim1 = (unsigned)(SNUM - kbase);  // rows available; OOB rows read 0

      u32x4 g0;
      g0[0] = 1u;                                   // count=1 (valid user D#)
      g0[1] = lds_addr;                             // lds_addr [63:32]
      g0[2] = (unsigned)(gaddr & 0xFFFFFFFFu);      // global_addr [95:64]
      g0[3] = (unsigned)((gaddr >> 32) & 0x1FFFFFFu) | (2u << 30);  // addr hi | type=2

      i32x8 g1;
      g1[0] = (int)(1u << 16);                      // data_size=1 (2-byte elements)
      g1[1] = (int)(64u << 16);                     // tensor_dim0=64 -> bits 63:48
      g1[2] = (int)(tdim1 << 16);                   // dim0 hi=0 | tensor_dim1 lo16
      g1[3] = (int)((tdim1 >> 16) | (64u << 16));   // tensor_dim1 hi | tile_dim0=64
      g1[4] = (int)32u;                             // tile_dim1=32, tile_dim2=0
      g1[5] = (int)512u;                            // tensor_dim0_stride=512 elems
      g1[6] = 0;                                    // stride hi | dim1_stride lo
      g1[7] = 0;

      i32x4 gz4 = {0, 0, 0, 0};                     // groups 2/3 unused (2D tensor)
      i32x8 gz8 = {0, 0, 0, 0, 0, 0, 0, 0};         // extra group (clang-23 6-arg form)
      __builtin_amdgcn_tensor_load_to_lds(g0, g1, gz4, gz4, gz8, 0);
    }
#else
    // ---- Fallback: manual staging; clamp OOB rows (P there is 0). ----
    {
      int vr = kbase + lane;
      if (vr > SNUM - 1) vr = SNUM - 1;
      const bf16_t* vrow = vh + (size_t)vr * DNUM;
#pragma unroll
      for (int c = 0; c < 64; c += 8)
        *(v8bf*)&vblk[lane][c] = *(const v8bf*)(vrow + c);
    }
#endif

    // Scores: two 16-key tiles; S = Q @ K^T  (WMMA K-dim = 64 features).
#pragma unroll
    for (int t = 0; t < 2; ++t) {
      int keyrow = kbase + t * 16 + (lane & 15);
      if (keyrow > SNUM - 1) keyrow = SNUM - 1;
      const bf16_t* krow = kh + (size_t)keyrow * DNUM + ((lane >> 4) * 16);
      v16bf kb0 = *(const v16bf*)(krow);       // features 0..31 chunk
      v16bf kb1 = *(const v16bf*)(krow + 32);  // features 32..63 chunk

      v8f s = {};
      s = wmma_bf16(qa0, kb0, s);
      s = wmma_bf16(qa1, kb1, s);

      int col = lane & 15;
      int rh  = (lane >> 4) * 8;
#pragma unroll
      for (int j = 0; j < 8; ++j) {
        int row = j + rh;
        int qg  = qt * 16 + row;
        int kg  = kbase + t * 16 + col;
        float val = s[j] * scale;
        if (kg > qg) val = -1e30f;
        sblk[row][t * 16 + col] = val;
      }
    }

    // Online softmax update (lanes 0..15, one query row each).
    if (lane < 16) {
      float m_old = mrow[lane];
      float bm = m_old;
#pragma unroll
      for (int c = 0; c < 32; ++c) bm = fmaxf(bm, sblk[lane][c]);
      float alpha = __expf(m_old - bm);
      float sum = 0.0f;
#pragma unroll
      for (int c = 0; c < 32; ++c) {
        float p = __expf(sblk[lane][c] - bm);
        sum += p;
        pblk[lane][c] = (bf16_t)p;
      }
      mrow[lane] = bm;
      lrow[lane] = lrow[lane] * alpha + sum;
      arow[lane] = alpha;
    }

    // Rescale running accumulator by per-row alpha.
    {
      int rh = (lane >> 4) * 8;
#pragma unroll
      for (int j = 0; j < 8; ++j) {
        float al = arow[j + rh];
#pragma unroll
        for (int nt = 0; nt < 4; ++nt) o[nt][j] *= al;
      }
    }

#if USE_TDM
    // V tile must be resident in LDS before we gather B operands from it.
    __builtin_amdgcn_s_wait_tensorcnt(0);
#endif

    // O += P(16x32) @ V(32x64): A from pblk, B gathered transposed from vblk.
    v16bf pa = load_a_tile(&pblk[0][0], 32, lane);
#pragma unroll
    for (int nt = 0; nt < 4; ++nt) {
      v16bf vb;
      int n   = lane & 15;
      int kk0 = (lane >> 4) * 16;
#pragma unroll
      for (int i = 0; i < 16; ++i)
        vb[i] = vblk[kk0 + i][nt * 16 + n];
      o[nt] = wmma_bf16(pa, vb, o[nt]);
    }
  }

  // Finalize: divide by row sums, write f32 attention output.
  int col = lane & 15;
  int rh  = (lane >> 4) * 8;
#pragma unroll
  for (int nt = 0; nt < 4; ++nt) {
#pragma unroll
    for (int j = 0; j < 8; ++j) {
      int row = j + rh;
      float invl = 1.0f / lrow[row];
      out[(size_t)(mbase + qt * 16 + row) * DNUM + h * DHNUM + nt * 16 + col] =
          o[nt][j] * invl;
    }
  }
}

// ---------------------------------------------------------------------------
// Fused residual-add + LayerNorm.  One 256-thread block per row (D=512).
// ---------------------------------------------------------------------------
__global__ void __launch_bounds__(256)
add_ln_kernel(const float* __restrict__ x, const float* __restrict__ res,
              const float* __restrict__ g, const float* __restrict__ bta,
              float* __restrict__ yf, bf16_t* __restrict__ yb) {
  __shared__ float red[256];
  int row = blockIdx.x;
  int t   = threadIdx.x;
  size_t o = (size_t)row * DNUM;

  float a0 = x[o + t];
  float a1 = x[o + t + 256];
  if (res) { a0 += res[o + t]; a1 += res[o + t + 256]; }

  red[t] = a0 + a1;
  __syncthreads();
  for (int off = 128; off > 0; off >>= 1) {
    if (t < off) red[t] += red[t + off];
    __syncthreads();
  }
  float mean = red[0] * (1.0f / DNUM);
  __syncthreads();

  float d0 = a0 - mean, d1 = a1 - mean;
  red[t] = d0 * d0 + d1 * d1;
  __syncthreads();
  for (int off = 128; off > 0; off >>= 1) {
    if (t < off) red[t] += red[t + off];
    __syncthreads();
  }
  float inv = rsqrtf(red[0] * (1.0f / DNUM) + 1e-5f);

  float y0 = d0 * inv * g[t]       + bta[t];
  float y1 = d1 * inv * g[t + 256] + bta[t + 256];
  if (yf) { yf[o + t] = y0; yf[o + t + 256] = y1; }
  yb[o + t]       = (bf16_t)y0;
  yb[o + t + 256] = (bf16_t)y1;
}

// ---------------------------------------------------------------------------
// Host-side orchestration
// ---------------------------------------------------------------------------
extern "C" void kernel_launch(void* const* d_in, const int* in_sizes, int n_in,
                              void* d_out, int out_size, void* d_ws, size_t ws_size,
                              hipStream_t stream) {
  (void)in_sizes; (void)n_in; (void)out_size; (void)ws_size;

  const int*   tokens = (const int*)  d_in[0];
  const float* emb    = (const float*)d_in[1];
  const float* Wq     = (const float*)d_in[2];
  const float* bq     = (const float*)d_in[3];
  const float* Wk     = (const float*)d_in[4];
  const float* bk     = (const float*)d_in[5];
  const float* Wv     = (const float*)d_in[6];
  const float* bv     = (const float*)d_in[7];
  const float* ln1_g  = (const float*)d_in[8];
  const float* ln1_b  = (const float*)d_in[9];
  const float* W1     = (const float*)d_in[10];
  const float* b1     = (const float*)d_in[11];
  const float* W2     = (const float*)d_in[12];
  const float* b2     = (const float*)d_in[13];
  const float* ln2_g  = (const float*)d_in[14];
  const float* ln2_b  = (const float*)d_in[15];
  const float* lnf_g  = (const float*)d_in[16];
  const float* lnf_b  = (const float*)d_in[17];
  const float* fc_W   = (const float*)d_in[18];
  const float* fc_b   = (const float*)d_in[19];
  float* out = (float*)d_out;

  char* ws = (char*)d_ws;
  size_t off = 0;
  auto wsalloc = [&](size_t bytes) -> void* {
    void* p = ws + off;
    off += (bytes + 255) & ~(size_t)255;
    return p;
  };

  float*  xf   = (float*) wsalloc((size_t)MNUM * DNUM * 4);
  bf16_t* xb   = (bf16_t*)wsalloc((size_t)MNUM * DNUM * 2);
  float*  tmpf = (float*) wsalloc((size_t)MNUM * DNUM * 4);
  bf16_t* qbuf = (bf16_t*)wsalloc((size_t)MNUM * DNUM * 2);
  bf16_t* kbuf = (bf16_t*)wsalloc((size_t)MNUM * DNUM * 2);
  bf16_t* vbuf = (bf16_t*)wsalloc((size_t)MNUM * DNUM * 2);
  bf16_t* hbuf = (bf16_t*)wsalloc((size_t)MNUM * FNUM * 2);
  bf16_t* wqt  = (bf16_t*)wsalloc((size_t)DNUM * DNUM * 2);
  bf16_t* wkt  = (bf16_t*)wsalloc((size_t)DNUM * DNUM * 2);
  bf16_t* wvt  = (bf16_t*)wsalloc((size_t)DNUM * DNUM * 2);
  bf16_t* w1t  = (bf16_t*)wsalloc((size_t)DNUM * FNUM * 2);
  bf16_t* w2t  = (bf16_t*)wsalloc((size_t)FNUM * DNUM * 2);
  bf16_t* fct  = (bf16_t*)wsalloc((size_t)DNUM * VNUM * 2);

  dim3 gblock(32, 4);

  // Embedding + PE.
  embed_kernel<<<(MNUM * DNUM) / 256, 256, 0, stream>>>(tokens, emb, xf, xb);

  // Vocabulary projection weights (independent, convert early).
  convert_wt_kernel<<<((size_t)DNUM * VNUM + 255) / 256, 256, 0, stream>>>(fc_W, fct, DNUM, VNUM);

  for (int l = 0; l < LNUM; ++l) {
    size_t wqo = (size_t)l * DNUM * DNUM;
    size_t w1o = (size_t)l * DNUM * FNUM;

    convert_wt_kernel<<<(DNUM * DNUM + 255) / 256, 256, 0, stream>>>(Wq + wqo, wqt, DNUM, DNUM);
    convert_wt_kernel<<<(DNUM * DNUM + 255) / 256, 256, 0, stream>>>(Wk + wqo, wkt, DNUM, DNUM);
    convert_wt_kernel<<<(DNUM * DNUM + 255) / 256, 256, 0, stream>>>(Wv + wqo, wvt, DNUM, DNUM);
    convert_wt_kernel<<<(DNUM * FNUM + 255) / 256, 256, 0, stream>>>(W1 + w1o, w1t, DNUM, FNUM);
    convert_wt_kernel<<<(FNUM * DNUM + 255) / 256, 256, 0, stream>>>(W2 + w1o, w2t, FNUM, DNUM);

    // QKV projections -> bf16.
    gemm_bf16_wmma<true, false><<<dim3(MNUM / 32, DNUM / 256), gblock, 0, stream>>>(
        xb, wqt, bq + (size_t)l * DNUM, nullptr, qbuf, DNUM, DNUM);
    gemm_bf16_wmma<true, false><<<dim3(MNUM / 32, DNUM / 256), gblock, 0, stream>>>(
        xb, wkt, bk + (size_t)l * DNUM, nullptr, kbuf, DNUM, DNUM);
    gemm_bf16_wmma<true, false><<<dim3(MNUM / 32, DNUM / 256), gblock, 0, stream>>>(
        xb, wvt, bv + (size_t)l * DNUM, nullptr, vbuf, DNUM, DNUM);

    // Causal attention -> f32 tmpf.
    attn_kernel<<<dim3(SNUM / 16, BNUM * HNUM), 32, 0, stream>>>(qbuf, kbuf, vbuf, tmpf);

    // x = LN(x + attn)
    add_ln_kernel<<<MNUM, 256, 0, stream>>>(xf, tmpf, ln1_g + (size_t)l * DNUM,
                                            ln1_b + (size_t)l * DNUM, xf, xb);

    // FFN: h = relu(x @ W1 + b1) (bf16), then tmpf = h @ W2 + b2 (f32).
    gemm_bf16_wmma<true, true><<<dim3(MNUM / 32, FNUM / 256), gblock, 0, stream>>>(
        xb, w1t, b1 + (size_t)l * FNUM, nullptr, hbuf, FNUM, DNUM);
    gemm_bf16_wmma<false, false><<<dim3(MNUM / 32, DNUM / 256), gblock, 0, stream>>>(
        hbuf, w2t, b2 + (size_t)l * DNUM, tmpf, nullptr, DNUM, FNUM);

    // x = LN(x + ffn)
    add_ln_kernel<<<MNUM, 256, 0, stream>>>(xf, tmpf, ln2_g + (size_t)l * DNUM,
                                            ln2_b + (size_t)l * DNUM, xf, xb);
  }

  // Final LayerNorm (bf16 only, feeds logits GEMM).
  add_ln_kernel<<<MNUM, 256, 0, stream>>>(xf, nullptr, lnf_g, lnf_b, nullptr, xb);

  // Logits: out[M, V] = xb @ fc_W + fc_b (f32 output straight to d_out).
  gemm_bf16_wmma<false, false><<<dim3(MNUM / 32, VNUM / 256), gblock, 0, stream>>>(
      xb, fct, fc_b, out, nullptr, VNUM, DNUM);
}
